// CausalMultiheadSelfAttention_56178172232146
// MI455X (gfx1250) — matrix operand order repaired and checked by reading, not compiled, not run on hardware
//
#include <hip/hip_runtime.h>

// ---------------------------------------------------------------------------
// Causal multi-head self-attention, MI455X (gfx1250, wave32, WMMA + TDM).
// bf16 operands + f32 accumulation via v_wmma_f32_16x16x32_bf16.
// Weight tiles staged into LDS by the Tensor Data Mover (double buffered).
// ---------------------------------------------------------------------------

#define D_MODEL   1024
#define NUM_HEADS 16
#define DK        64
#define BATCH     4
#define SEQ       2048
#define MROWS     (BATCH * SEQ)   // 8192

// TDM-staged weight tile: 64 rows x 128B (64 bf16), padded 16B per 128B row.
#define WTILE_ROWS   64
#define WTILE_KB     64            // K elements per stage
#define LDS_PITCH    144           // 128 + 16 pad (9*16 -> keeps 16B alignment)
#define LDS_BUF_SZ   (WTILE_ROWS * LDS_PITCH)   // 9216 B
#define LDS_TOTAL    (2 * LDS_BUF_SZ)           // 18432 B

typedef __bf16 bf16_t;
typedef __bf16        v16bf __attribute__((ext_vector_type(16)));
typedef __bf16        v8bf  __attribute__((ext_vector_type(8)));
typedef float         v8f   __attribute__((ext_vector_type(8)));
typedef float         v4f   __attribute__((ext_vector_type(4)));
typedef unsigned int  v4u   __attribute__((ext_vector_type(4)));
typedef unsigned int  u32x4 __attribute__((ext_vector_type(4)));
typedef int           i32x8 __attribute__((ext_vector_type(8)));
typedef int           i32x4 __attribute__((ext_vector_type(4)));

struct FragBits { v4u lo, hi; };

// A-matrix fragment (16x32 bf16, MxK): lane m = lane&15.
// lane-group 0 holds K = {k0..k0+7, k0+16..k0+23}, group 1 = {+8 of each}.
// Caller passes p = rowbase + k0 + (lane>>4)*8.
__device__ __forceinline__ v16bf load_frag_a(const bf16_t* p) {
  FragBits t;
  t.lo = *reinterpret_cast<const v4u*>(p);
  t.hi = *reinterpret_cast<const v4u*>(p + 16);
  return __builtin_bit_cast(v16bf, t);
}

// B-matrix fragment (32x16 bf16, KxN): lane n = lane&15.
// lane-group 0 holds K = {k0..k0+7, k0+16..k0+23}, group 1 = {+8 of each}.
// Caller passes p = colbase + k0 + (lane>>4)*8 (data along K contiguous).
__device__ __forceinline__ v16bf load_frag_b(const bf16_t* p) {
  FragBits t;
  t.lo = *reinterpret_cast<const v4u*>(p);
  t.hi = *reinterpret_cast<const v4u*>(p + 16);
  return __builtin_bit_cast(v16bf, t);
}

__device__ __forceinline__ v8f wmma_bf16(v16bf a, v16bf b, v8f c) {
  return __builtin_amdgcn_wmma_f32_16x16x32_bf16(
      /*neg_a=*/false, a, /*neg_b=*/false, b,
      /*c_mod=*/(short)0, c, /*reuse_a=*/false, /*reuse_b=*/false);
}

__device__ __forceinline__ v8f zero8() {
  v8f z;
#pragma unroll
  for (int i = 0; i < 8; ++i) z[i] = 0.0f;
  return z;
}

// ---------------------------------------------------------------------------
// TDM: DMA one 64-row x 128-byte weight tile (2D) from global into LDS,
// 16B of LDS padding per 128B row (bank-conflict-free fragment reads).
// D# layout per CDNA5 ISA ch.8 (group0: count/lds/global/type, group1: dims).
// 6-arg builtin form (amdgpu-toolchain): (g0, g1, g2, g3, g4, cpol).
// ---------------------------------------------------------------------------
__device__ __forceinline__ void tdm_load_wtile(const bf16_t* gptr, unsigned lds_off) {
  unsigned long long ga = (unsigned long long)(uintptr_t)gptr;
  u32x4 g0;
  g0[0] = 1u;                                    // count=1 (user descriptor)
  g0[1] = lds_off;                               // lds_addr (bytes)
  g0[2] = (unsigned)(ga & 0xFFFFFFFFu);          // global_addr[31:0]
  g0[3] = ((unsigned)(ga >> 32) & 0x01FFFFFFu)   // global_addr[56:32]
          | 0x80000000u;                         // type=2 @ bits[127:126]
  i32x8 g1;
  g1[0] = (3 << 16)      // data_size = 8B units
        | (1 << 20)      // pad_enable
        | (4 << 22)      // pad_interval: 32 DWORDs = 128B
        | (3 << 25);     // pad_amount : 4 DWORDs = 16B
  g1[1] = (int)(256u  << 16);   // tensor_dim0 = 256 (x8B = 2048B row)
  g1[2] = (int)(1024u << 16);   // tensor_dim1 = 1024 rows
  g1[3] = (int)(16u   << 16);   // tile_dim0 = 16 (x8B = 128B)
  g1[4] = WTILE_ROWS;           // tile_dim1 = 64, tile_dim2 = 0
  g1[5] = 256;                  // tensor_dim0_stride = 256 (x8B = 2048B)
  g1[6] = 0;
  g1[7] = 0;
  i32x4 z4;
  z4[0] = 0; z4[1] = 0; z4[2] = 0; z4[3] = 0;
  i32x8 z8;
#pragma unroll
  for (int i = 0; i < 8; ++i) z8[i] = 0;
  __builtin_amdgcn_tensor_load_to_lds(g0, g1, z4, z4, z8, 0);
}

// ---------------------------------------------------------------------------
// fp32 -> bf16 streaming convert (n multiple of 8)
// ---------------------------------------------------------------------------
__global__ __launch_bounds__(256)
void cvt_f32_to_bf16(const float* __restrict__ src, bf16_t* __restrict__ dst, int n) {
  int i = (blockIdx.x * blockDim.x + threadIdx.x) * 8;
  if (i >= n) return;
  v4f a = *reinterpret_cast<const v4f*>(src + i);
  v4f b = *reinterpret_cast<const v4f*>(src + i + 4);
  v8bf o;
#pragma unroll
  for (int j = 0; j < 4; ++j) {
    o[j]     = (bf16_t)a[j];
    o[j + 4] = (bf16_t)b[j];
  }
  *reinterpret_cast<v8bf*>(dst + i) = o;
}

// ---------------------------------------------------------------------------
// GEMM: C[M=8192, N=1024] = A[M,K=1024] * W[N,K]^T  (Linear: x @ W.T)
// Block = 128 threads = 4 waves; wave computes 32x64 of C (2 Mtiles x 4 Ntiles).
// W tile (64 rows x 64 K) staged in LDS by TDM, double-buffered; A direct.
// mode 0: Q -> bf16 [B,H,S,dk], scaled by 1/sqrt(dk)
// mode 1: K -> bf16 [B,H,S,dk]
// mode 2: V -> bf16 [B,H,dk,S]   (transposed for attention B-fragments)
// mode 3: O -> fp32 [M,N] (final output)
// ---------------------------------------------------------------------------
__global__ __launch_bounds__(128)
void gemm_bf16(const bf16_t* __restrict__ A, const bf16_t* __restrict__ W,
               bf16_t* __restrict__ dstb, float* __restrict__ dstf, int mode) {
  extern __shared__ char smem[];   // 2 x 9216 B, TDM target (offset 0)

  const int lane = threadIdx.x & 31;
  const int wave = threadIdx.x >> 5;
  const int row  = lane & 15;
  const int half = lane >> 4;
  const int m0 = blockIdx.y * 128 + wave * 32;
  const int n0 = blockIdx.x * 64;

  v8f acc[2][4];
#pragma unroll
  for (int mt = 0; mt < 2; ++mt)
#pragma unroll
    for (int nt = 0; nt < 4; ++nt) acc[mt][nt] = zero8();

  const bf16_t* a0p = A + (size_t)(m0 + row) * D_MODEL + half * 8;
  const bf16_t* a1p = A + (size_t)(m0 + 16 + row) * D_MODEL + half * 8;
  const bf16_t* wtile0 = W + (size_t)n0 * D_MODEL;   // row n0, K chunk advances

  const int NSTAGE = D_MODEL / WTILE_KB;   // 16

  // Prologue: stage 0 into buffer 0.
  if (wave == 0) tdm_load_wtile(wtile0, 0);

  for (int i = 0; i < NSTAGE; ++i) {
    // (a) all waves finished reading the buffer we are about to overwrite
    __syncthreads();
    if (wave == 0) {
      if (i + 1 < NSTAGE) {
        tdm_load_wtile(wtile0 + (i + 1) * WTILE_KB,
                       (unsigned)(((i + 1) & 1) * LDS_BUF_SZ));
        __builtin_amdgcn_s_wait_tensorcnt(1);   // stage i complete (in-order)
      } else {
        __builtin_amdgcn_s_wait_tensorcnt(0);
      }
    }
    // (b) broadcast "stage i data ready" to all waves
    __syncthreads();

    const char* buf = smem + (i & 1) * LDS_BUF_SZ;
#pragma unroll
    for (int kk = 0; kk < 2; ++kk) {                 // two 32-wide WMMA K steps
      const int k0 = i * WTILE_KB + kk * 32;
      v16bf a0 = load_frag_a(a0p + k0);
      v16bf a1 = load_frag_a(a1p + k0);
      // Preload all 4 B-fragments into distinct registers so the 8
      // ds_load_b128s issue as one clause and overlap the WMMA burst.
      v16bf bfr[4];
#pragma unroll
      for (int nt = 0; nt < 4; ++nt) {
        const bf16_t* bp = reinterpret_cast<const bf16_t*>(
            buf + (nt * 16 + row) * LDS_PITCH + kk * 64 + half * 16);
        bfr[nt] = load_frag_b(bp);
      }
#pragma unroll
      for (int nt = 0; nt < 4; ++nt) {
        acc[0][nt] = wmma_bf16(a0, bfr[nt], acc[0][nt]);
        acc[1][nt] = wmma_bf16(a1, bfr[nt], acc[1][nt]);
      }
    }
  }

  // Epilogue: C layout -> VGPR j holds (m = j + half*8, n = lane&15) per tile.
#pragma unroll
  for (int mt = 0; mt < 2; ++mt) {
#pragma unroll
    for (int nt = 0; nt < 4; ++nt) {
#pragma unroll
      for (int j = 0; j < 8; ++j) {
        int m = m0 + mt * 16 + half * 8 + j;
        int n = n0 + nt * 16 + row;
        float v = acc[mt][nt][j];
        if (mode == 3) {
          dstf[(size_t)m * D_MODEL + n] = v;
        } else {
          int b = m >> 11, s = m & (SEQ - 1);
          int h = n >> 6,  d = n & (DK - 1);
          if (mode == 2) {
            dstb[(((size_t)(b * NUM_HEADS + h) * DK + d) * SEQ) + s] = (bf16_t)v;
          } else {
            float sc = (mode == 0) ? 0.125f : 1.0f;  // 1/sqrt(dk)
            dstb[(((size_t)(b * NUM_HEADS + h) * SEQ + s) * DK) + d] = (bf16_t)(v * sc);
          }
        }
      }
    }
  }
}

// ---------------------------------------------------------------------------
// Flash attention (causal), one 16-query tile per wave.
// Scores computed transposed: S^T = K * Q^T  -> each lane owns one query row;
// softmax is in-lane + one shfl_xor(16). The exp'd score C-tiles ARE the
// documented A-fragment layout for P, so P*V needs no LDS / transpose.
// ---------------------------------------------------------------------------
__global__ __launch_bounds__(128)
void attn_causal(const bf16_t* __restrict__ Qb, const bf16_t* __restrict__ Kb,
                 const bf16_t* __restrict__ Vt, bf16_t* __restrict__ AO) {
  const int lane = threadIdx.x & 31;
  const int wave = threadIdx.x >> 5;
  const int row  = lane & 15;
  const int half = lane >> 4;
  const int bh = blockIdx.y;               // 0..63
  const int q0 = (blockIdx.x * 4 + wave) * 16;
  const int bb = bh / NUM_HEADS, hh = bh % NUM_HEADS;

  const bf16_t* Qh = Qb + (size_t)bh * SEQ * DK;
  const bf16_t* Kh = Kb + (size_t)bh * SEQ * DK;
  const bf16_t* Vh = Vt + (size_t)bh * DK * SEQ;

  // Q^T B-fragments: lane n = query, K dim = dk (2 chunks of 32).
  v16bf qf0 = load_frag_b(Qh + (size_t)(q0 + row) * DK + 0  + half * 8);
  v16bf qf1 = load_frag_b(Qh + (size_t)(q0 + row) * DK + 32 + half * 8);

  v8f o[4];
#pragma unroll
  for (int t = 0; t < 4; ++t) o[t] = zero8();
  float m_run = -1e30f, l_run = 0.0f;
  const int qg = q0 + row;                 // query row owned by this lane
  const int kmax = q0 + 16;                // keys 0..q0+15 needed (causal)

  for (int kb = 0; kb < kmax; kb += 32) {
    // S^T tiles: t0 = keys [kb, kb+16), t1 = keys [kb+16, kb+32)
    v8f s0 = zero8(), s1 = zero8();
    v16bf ka0 = load_frag_a(Kh + (size_t)(kb + row) * DK + 0  + half * 8);
    v16bf ka1 = load_frag_a(Kh + (size_t)(kb + row) * DK + 32 + half * 8);
    v16bf ka2 = load_frag_a(Kh + (size_t)(kb + 16 + row) * DK + 0  + half * 8);
    v16bf ka3 = load_frag_a(Kh + (size_t)(kb + 16 + row) * DK + 32 + half * 8);
    s0 = wmma_bf16(ka0, qf0, s0);
    s0 = wmma_bf16(ka1, qf1, s0);
    s1 = wmma_bf16(ka2, qf0, s1);
    s1 = wmma_bf16(ka3, qf1, s1);

    // Causal mask + per-lane block max.  C layout: VGPR j -> key kb + half*8 + j.
    float bm = -1e30f;
#pragma unroll
    for (int j = 0; j < 8; ++j) {
      int key0 = kb + half * 8 + j;
      float x0 = (key0 <= qg) ? s0[j] : -1e30f;
      float x1 = (key0 + 16 <= qg) ? s1[j] : -1e30f;
      s0[j] = x0; s1[j] = x1;
      bm = fmaxf(bm, fmaxf(x0, x1));
    }
    bm = fmaxf(bm, __shfl_xor(bm, 16, 32));
    float m_new   = fmaxf(m_run, bm);
    float rescale = __expf(m_run - m_new);

    float p0[8], p1[8], psum = 0.0f;
#pragma unroll
    for (int j = 0; j < 8; ++j) {
      p0[j] = __expf(s0[j] - m_new);
      p1[j] = __expf(s1[j] - m_new);
      psum += p0[j] + p1[j];
    }
    psum += __shfl_xor(psum, 16, 32);
    l_run = l_run * rescale + psum;
    m_run = m_new;

    // Rescale O: O tile has query on VGPR rows -> broadcast per-query factor.
#pragma unroll
    for (int j = 0; j < 8; ++j) {
      float rj = __shfl(rescale, (half << 3) + j, 32);
      o[0][j] *= rj; o[1][j] *= rj; o[2][j] *= rj; o[3][j] *= rj;
    }

    // P A-fragment straight from the score tiles (C layout == A layout here).
    v16bf pf;
#pragma unroll
    for (int j = 0; j < 8; ++j) {
      pf[j]     = (bf16_t)p0[j];
      pf[j + 8] = (bf16_t)p1[j];
    }

    // O += P * V;  V^T gives contiguous per-lane B-fragment loads.
    v16bf vbs[4];
#pragma unroll
    for (int dt = 0; dt < 4; ++dt)
      vbs[dt] = load_frag_b(Vh + (size_t)(dt * 16 + row) * SEQ + kb + half * 8);
#pragma unroll
    for (int dt = 0; dt < 4; ++dt)
      o[dt] = wmma_bf16(pf, vbs[dt], o[dt]);

    if (kb + 32 < kmax) {
      __builtin_prefetch(Kh + (size_t)(kb + 32 + row) * DK, 0, 1);
      __builtin_prefetch(Vh + (size_t)row * SEQ + kb + 32, 0, 1);
    }
  }

  float inv = 1.0f / l_run;
  // Store AO[b][s][h*64 + d] as bf16 for the O-projection GEMM.
#pragma unroll
  for (int j = 0; j < 8; ++j) {
    float ij = __shfl(inv, (half << 3) + j, 32);
    int s = q0 + half * 8 + j;
    size_t base = ((size_t)(bb * SEQ + s)) * D_MODEL + hh * DK + row;
#pragma unroll
    for (int dt = 0; dt < 4; ++dt) {
      AO[base + dt * 16] = (bf16_t)(o[dt][j] * ij);
    }
  }
}

// ---------------------------------------------------------------------------
// Host-side launch
// ---------------------------------------------------------------------------
extern "C" void kernel_launch(void* const* d_in, const int* in_sizes, int n_in,
                              void* d_out, int out_size, void* d_ws, size_t ws_size,
                              hipStream_t stream) {
  (void)in_sizes; (void)n_in; (void)out_size; (void)ws_size;

  const float* x  = (const float*)d_in[0];
  const float* Wq = (const float*)d_in[1];
  const float* Wk = (const float*)d_in[2];
  const float* Wv = (const float*)d_in[3];
  const float* Wo = (const float*)d_in[4];
  float* out = (float*)d_out;

  const size_t NX = (size_t)MROWS * D_MODEL;     // 8M elements
  const size_t NW = (size_t)D_MODEL * D_MODEL;   // 1M elements

  bf16_t* xb  = (bf16_t*)d_ws;
  bf16_t* wqb = xb  + NX;
  bf16_t* wkb = wqb + NW;
  bf16_t* wvb = wkb + NW;
  bf16_t* wob = wvb + NW;
  bf16_t* Qb  = wob + NW;   // [B,H,S,dk] bf16, pre-scaled by 1/sqrt(dk)
  bf16_t* Kb  = Qb  + NX;   // [B,H,S,dk]
  bf16_t* Vt  = Kb  + NX;   // [B,H,dk,S]
  bf16_t* AO  = Vt  + NX;   // [B,S,D] attention output

  cvt_f32_to_bf16<<<(int)(NX / 2048), 256, 0, stream>>>(x,  xb,  (int)NX);
  cvt_f32_to_bf16<<<(int)(NW / 2048), 256, 0, stream>>>(Wq, wqb, (int)NW);
  cvt_f32_to_bf16<<<(int)(NW / 2048), 256, 0, stream>>>(Wk, wkb, (int)NW);
  cvt_f32_to_bf16<<<(int)(NW / 2048), 256, 0, stream>>>(Wv, wvb, (int)NW);
  cvt_f32_to_bf16<<<(int)(NW / 2048), 256, 0, stream>>>(Wo, wob, (int)NW);

  dim3 gg(D_MODEL / 64, MROWS / 128);   // (16, 64), 128 threads
  gemm_bf16<<<gg, 128, LDS_TOTAL, stream>>>(xb, wqb, Qb, nullptr, 0);
  gemm_bf16<<<gg, 128, LDS_TOTAL, stream>>>(xb, wkb, Kb, nullptr, 1);
  gemm_bf16<<<gg, 128, LDS_TOTAL, stream>>>(xb, wvb, Vt, nullptr, 2);

  dim3 ga(SEQ / 64, BATCH * NUM_HEADS); // (32, 64), 4 query-tiles per block
  attn_causal<<<ga, 128, 0, stream>>>(Qb, Kb, Vt, AO);

  gemm_bf16<<<gg, 128, LDS_TOTAL, stream>>>(AO, wob, nullptr, out, 3);
}
